// CrossAttention_47356309406304
// MI455X (gfx1250) — matrix operand order repaired and checked by reading, not compiled, not run on hardware
//
#include <hip/hip_runtime.h>

// ---------------------------------------------------------------------------
// CrossAttention for MI455X (gfx1250, wave32, WMMA 16x16x32 bf16)
// B=8, T=512, N=1024, Df=1024, H=16, HD=64
// - GEMMs: 32-row M-tiles, every weight B-frag feeds 2 WMMAs (~51 FLOP/B L2)
// - Attention: K/V^T chunks staged to LDS with global_load_async_to_lds_b128
//   (double-buffered, s_wait_asynccnt), shared by all 4 waves of the block.
// ---------------------------------------------------------------------------

#define Bsz 8
#define Tt 512
#define Nn 1024
#define Df 1024
#define Hh 16
#define HD 64

typedef __attribute__((ext_vector_type(16))) __bf16 v16bf;
typedef __attribute__((ext_vector_type(8)))  float  v8f;

union Frag {
  v16bf v;
  unsigned short u[16];
  uint4 q[2];
};

__device__ __forceinline__ unsigned short f2bf(float f) {
  unsigned int u = __float_as_uint(f);
  return (unsigned short)((u + 0x7FFFu + ((u >> 16) & 1u)) >> 16);
}

__device__ __forceinline__ v8f wmma_bf16(v16bf a, v16bf b, v8f c) {
  return __builtin_amdgcn_wmma_f32_16x16x32_bf16(false, a, false, b, (short)0, c, false, false);
}

// Async copy of 16 bytes global -> LDS (gfx1250, tracked by ASYNCcnt).
__device__ __forceinline__ void async_ld16(unsigned lds_off, const void* gptr) {
  asm volatile("global_load_async_to_lds_b128 %0, %1, off"
               :: "v"(lds_off), "v"((unsigned long long)gptr)
               : "memory");
}

// A-fragment (16x32, rows row0..row0+15, K = k0..k0+31), row-major source.
// lane<16: row=lane, K {0..7,16..23}; lane>=16: row=lane-16, K {8..15,24..31}
__device__ __forceinline__ v16bf ld_afrag(const unsigned short* base, int row0,
                                          int ld, int k0) {
  const int lane = threadIdx.x & 31;
  const unsigned short* p =
      base + (size_t)(row0 + (lane & 15)) * ld + k0 + ((lane >> 4) << 3);
  Frag f;
  f.q[0] = *(const uint4*)(p);
  f.q[1] = *(const uint4*)(p + 16);
  return f.v;
}

// B-fragment (32x16): column n = col0 + lane%16 taken as a row of the
// transposed operand (row-major along K). lane<16: K {0..7,16..23}; lane>=16: K {8..15,24..31}
__device__ __forceinline__ v16bf ld_bfrag(const unsigned short* base, int col0,
                                          int ld, int k0) {
  const int lane = threadIdx.x & 31;
  const unsigned short* p =
      base + (size_t)(col0 + (lane & 15)) * ld + k0 + ((lane >> 4) << 3);
  Frag f;
  f.q[0] = *(const uint4*)(p);
  f.q[1] = *(const uint4*)(p + 16);
  return f.v;
}

// --------------------------- fp32 -> bf16 convert ---------------------------
__global__ void cvt_bf16(const float* __restrict__ in,
                         unsigned short* __restrict__ out, int n) {
  int i = blockIdx.x * blockDim.x + threadIdx.x;
  if (i < n) out[i] = f2bf(in[i]);
}

// ------------------ RMSNorm(features) -> feats_bf16 + V^T -------------------
__global__ __launch_bounds__(256) void rmsnorm_feat(
    const float* __restrict__ X, const float* __restrict__ g,
    unsigned short* __restrict__ feats, unsigned short* __restrict__ vt) {
  const int row = blockIdx.x;  // b*N + n
  const int b = row >> 10, n = row & 1023;
  const float* x = X + (size_t)row * Df;
  float v[4];
  float s = 0.f;
#pragma unroll
  for (int i = 0; i < 4; ++i) {
    v[i] = x[threadIdx.x + i * 256];
    s += v[i] * v[i];
  }
#pragma unroll
  for (int off = 16; off >= 1; off >>= 1) s += __shfl_xor(s, off, 32);
  __shared__ float wsum[8];
  const int wid = threadIdx.x >> 5, lane = threadIdx.x & 31;
  if (lane == 0) wsum[wid] = s;
  __syncthreads();
  float tot = 0.f;
#pragma unroll
  for (int i = 0; i < 8; ++i) tot += wsum[i];
  const float inv = __frsqrt_rn(tot * (1.f / Df) + 1e-6f);
  unsigned short* fr = feats + (size_t)row * Df;
  unsigned short* vtb = vt + (size_t)b * Df * Nn;
#pragma unroll
  for (int i = 0; i < 4; ++i) {
    int d = threadIdx.x + i * 256;
    unsigned short hv = f2bf(v[i] * inv * g[d]);
    fr[d] = hv;                        // [b,n,d] row-major
    vtb[(size_t)d * Nn + n] = hv;      // [b,d,n] transposed for P@V B-frags
  }
}

// ---------- GEMM (32-row strip x 1024 cols) + bias + RMSNorm -> bf16 --------
// out[m,n] = rmsnorm_row(A[m,:] . W[n,:] + bias[n]) * g[n]
__global__ __launch_bounds__(256) void gemm16_rms(
    const unsigned short* __restrict__ A, const unsigned short* __restrict__ W,
    const float* __restrict__ bias, const float* __restrict__ g,
    unsigned short* __restrict__ out) {
  const int row0 = blockIdx.x * 32;
  const int wid = threadIdx.x >> 5;
  const int lane = threadIdx.x & 31;
  const int colbase = wid * 128;  // 8 waves x 128 cols = 1024

  v8f acc[2][8];
  const v8f vzero = {0.f, 0.f, 0.f, 0.f, 0.f, 0.f, 0.f, 0.f};
#pragma unroll
  for (int m = 0; m < 2; ++m)
#pragma unroll
    for (int j = 0; j < 8; ++j) acc[m][j] = vzero;

  for (int k = 0; k < Df; k += 32) {
    v16bf a0 = ld_afrag(A, row0, Df, k);
    v16bf a1 = ld_afrag(A, row0 + 16, Df, k);
#pragma unroll
    for (int j = 0; j < 8; ++j) {
      v16bf b = ld_bfrag(W, colbase + j * 16, Df, k);
      acc[0][j] = wmma_bf16(a0, b, acc[0][j]);
      acc[1][j] = wmma_bf16(a1, b, acc[1][j]);
    }
  }

  __shared__ float rowsq[32];
  if (threadIdx.x < 32) rowsq[threadIdx.x] = 0.f;
  __syncthreads();

  const int colLane = lane & 15;
  const int rbase = (lane < 16) ? 0 : 8;
#pragma unroll
  for (int m = 0; m < 2; ++m) {
    float sq[8];
#pragma unroll
    for (int e = 0; e < 8; ++e) sq[e] = 0.f;
#pragma unroll
    for (int j = 0; j < 8; ++j) {
      float bcol = bias[colbase + j * 16 + colLane];
#pragma unroll
      for (int e = 0; e < 8; ++e) {
        float vv = acc[m][j][e] + bcol;
        acc[m][j][e] = vv;
        sq[e] += vv * vv;
      }
    }
#pragma unroll
    for (int e = 0; e < 8; ++e) atomicAdd(&rowsq[m * 16 + rbase + e], sq[e]);
  }
  __syncthreads();

#pragma unroll
  for (int m = 0; m < 2; ++m) {
#pragma unroll
    for (int e = 0; e < 8; ++e) {
      float inv = __frsqrt_rn(rowsq[m * 16 + rbase + e] * (1.f / Df) + 1e-6f);
      int row = row0 + m * 16 + rbase + e;
#pragma unroll
      for (int j = 0; j < 8; ++j) {
        int col = colbase + j * 16 + colLane;
        out[(size_t)row * Df + col] = f2bf(acc[m][j][e] * inv * g[col]);
      }
    }
  }
}

// ------------------- GEMM + bias -> fp32 (output projection) ----------------
__global__ __launch_bounds__(256) void gemm16_bias(
    const unsigned short* __restrict__ A, const unsigned short* __restrict__ W,
    const float* __restrict__ bias, float* __restrict__ out) {
  const int row0 = blockIdx.x * 32;
  const int wid = threadIdx.x >> 5;
  const int lane = threadIdx.x & 31;
  const int colbase = wid * 128;

  v8f acc[2][8];
  const v8f vzero = {0.f, 0.f, 0.f, 0.f, 0.f, 0.f, 0.f, 0.f};
#pragma unroll
  for (int m = 0; m < 2; ++m)
#pragma unroll
    for (int j = 0; j < 8; ++j) acc[m][j] = vzero;

  for (int k = 0; k < Df; k += 32) {
    v16bf a0 = ld_afrag(A, row0, Df, k);
    v16bf a1 = ld_afrag(A, row0 + 16, Df, k);
#pragma unroll
    for (int j = 0; j < 8; ++j) {
      v16bf b = ld_bfrag(W, colbase + j * 16, Df, k);
      acc[0][j] = wmma_bf16(a0, b, acc[0][j]);
      acc[1][j] = wmma_bf16(a1, b, acc[1][j]);
    }
  }

  const int colLane = lane & 15;
  const int rbase = (lane < 16) ? 0 : 8;
#pragma unroll
  for (int m = 0; m < 2; ++m) {
#pragma unroll
    for (int e = 0; e < 8; ++e) {
      int row = row0 + m * 16 + rbase + e;
#pragma unroll
      for (int j = 0; j < 8; ++j) {
        int col = colbase + j * 16 + colLane;
        out[(size_t)row * Df + col] = acc[m][j][e] + bias[col];
      }
    }
  }
}

// ----------------------------- Flash attention ------------------------------
// Block = (b, h, 64 queries), 4 waves x 16-query tiles. K/V^T chunks for each
// 32-kv step are staged once per block into LDS via async-to-LDS loads
// (double-buffered, ASYNCcnt), then all 4 waves read frags via ds_load_b128.
// Scores computed transposed (S^T = K.Q^T) so the softmaxed accumulator is
// already in A-frag layout for P @ V (V supplied transposed).
__global__ __launch_bounds__(128) void attention(
    const unsigned short* __restrict__ Q, const unsigned short* __restrict__ K,
    const unsigned short* __restrict__ Vt, unsigned short* __restrict__ AO) {
  __shared__ unsigned short kbuf[2][32 * 64];  // [kv 0..31][d 0..63]
  __shared__ unsigned short vbuf[2][64 * 32];  // [d 0..63][kv 0..31]

  const int b = blockIdx.z, h = blockIdx.y;
  const int wid = threadIdx.x >> 5, lane = threadIdx.x & 31;
  const int tid = threadIdx.x;
  const int t0 = blockIdx.x * 64 + wid * 16;
  const int d0 = h * HD;
  const unsigned short* Qb = Q + (size_t)b * Tt * Df;
  const unsigned short* Kb = K + (size_t)b * Nn * Df + d0;       // cols folded
  const unsigned short* Vb = Vt + (size_t)b * Df * Nn + (size_t)d0 * Nn;

  // stage K[n0..n0+31][d0..d0+63] and V^T[d0..d0+63][n0..n0+31] into buffer
  auto stage = [&](int buf, int n0) {
#pragma unroll
    for (int r = 0; r < 2; ++r) {
      int chunk = tid + r * 128;  // 256 x 16B = 4KB K tile rows of 128B
      int row = chunk >> 3, c = chunk & 7;
      async_ld16((unsigned)(size_t)&kbuf[buf][chunk * 8],
                 Kb + (size_t)(n0 + row) * Df + c * 8);
    }
#pragma unroll
    for (int r = 0; r < 2; ++r) {
      int chunk = tid + r * 128;  // V^T tile rows of 64B
      int row = chunk >> 2, c = chunk & 3;
      async_ld16((unsigned)(size_t)&vbuf[buf][chunk * 8],
                 Vb + (size_t)row * Nn + n0 + c * 8);
    }
  };

  const v16bf qb0 = ld_bfrag(Qb, t0, Df, d0);       // Q^T B-frags (d 0..31)
  const v16bf qb1 = ld_bfrag(Qb, t0, Df, d0 + 32);  // (d 32..63)

  const v8f vzero = {0.f, 0.f, 0.f, 0.f, 0.f, 0.f, 0.f, 0.f};
  v8f o[4];
#pragma unroll
  for (int j = 0; j < 4; ++j) o[j] = vzero;
  float m = -1e30f, l = 0.f;  // lane owns query q = lane%16 (dup in halves)

  stage(0, 0);
  for (int it = 0; it < Nn / 32; ++it) {
    const int buf = it & 1;
    if (it + 1 < Nn / 32) {
      stage(buf ^ 1, (it + 1) * 32);
      // 4 newer async loads outstanding; wait for the 4 of stage `it`
      asm volatile("s_wait_asynccnt 0x4" ::: "memory");
    } else {
      asm volatile("s_wait_asynccnt 0x0" ::: "memory");
    }
    __syncthreads();  // all waves' copies of stage `it` visible

    const unsigned short* kb = &kbuf[buf][0];
    const unsigned short* vb = &vbuf[buf][0];

    // S^T tiles: [kv 0..15, q] and [kv 16..31, q], K-dim = 64 (2 wmma each)
    v8f s0 = wmma_bf16(ld_afrag(kb, 0, 64, 0), qb0, vzero);
    s0 = wmma_bf16(ld_afrag(kb, 0, 64, 32), qb1, s0);
    v8f s1 = wmma_bf16(ld_afrag(kb, 16, 64, 0), qb0, vzero);
    s1 = wmma_bf16(ld_afrag(kb, 16, 64, 32), qb1, s1);

    float p0[8], p1[8], cm = -1e30f;
#pragma unroll
    for (int e = 0; e < 8; ++e) {
      p0[e] = s0[e] * 0.125f;  // 1/sqrt(64)
      p1[e] = s1[e] * 0.125f;
      cm = fmaxf(cm, fmaxf(p0[e], p1[e]));
    }
    cm = fmaxf(cm, __shfl_xor(cm, 16, 32));  // combine kv halves
    const float mnew = fmaxf(m, cm);
    const float corr = __expf(m - mnew);

    Frag pf;
    float rs = 0.f;
#pragma unroll
    for (int e = 0; e < 8; ++e) {
      float e0 = __expf(p0[e] - mnew);
      float e1 = __expf(p1[e] - mnew);
      rs += e0 + e1;
      pf.u[e] = f2bf(e0);      // kv {0..7}/{8..15}  -> A-frag K low half
      pf.u[8 + e] = f2bf(e1);  // kv {16..23}/{24..31} -> A-frag K high half
    }
    rs += __shfl_xor(rs, 16, 32);
    l = l * corr + rs;
    m = mnew;

    // rescale running O: rows of O live on VGPR index, corr lives on lane=q
    float corr_e[8];
#pragma unroll
    for (int e = 0; e < 8; ++e)
      corr_e[e] = __shfl(corr, (lane < 16) ? e : e + 8, 32);
#pragma unroll
    for (int j = 0; j < 4; ++j) {
#pragma unroll
      for (int e = 0; e < 8; ++e) o[j][e] *= corr_e[e];
    }

    // O += P @ V_chunk  (V^T rows are contiguous B-frags in LDS)
#pragma unroll
    for (int j = 0; j < 4; ++j) {
      v16bf vfr = ld_bfrag(vb, j * 16, 32, 0);
      o[j] = wmma_bf16(pf.v, vfr, o[j]);
    }
    __syncthreads();  // done reading `buf` before it is restaged next iter
  }

  // normalize by l and store bf16 for the output projection
#pragma unroll
  for (int e = 0; e < 8; ++e) {
    int r = (lane < 16) ? e : e + 8;
    float linv = 1.f / __shfl(l, r, 32);
    int row = t0 + r;
#pragma unroll
    for (int j = 0; j < 4; ++j) {
      int col = d0 + j * 16 + (lane & 15);
      AO[((size_t)b * Tt + row) * Df + col] = f2bf(o[j][e] * linv);
    }
  }
}

// ---------------------------------------------------------------------------
extern "C" void kernel_launch(void* const* d_in, const int* in_sizes, int n_in,
                              void* d_out, int out_size, void* d_ws,
                              size_t ws_size, hipStream_t stream) {
  const float* text = (const float*)d_in[0];
  const float* feat = (const float*)d_in[1];
  const float* W_q = (const float*)d_in[2];
  const float* b_q = (const float*)d_in[3];
  const float* W_k = (const float*)d_in[4];
  const float* b_k = (const float*)d_in[5];
  const float* W_o = (const float*)d_in[6];
  const float* b_o = (const float*)d_in[7];
  const float* g_feat = (const float*)d_in[8];
  const float* g_q = (const float*)d_in[9];
  const float* g_k = (const float*)d_in[10];
  float* out = (float*)d_out;

  unsigned short* p = (unsigned short*)d_ws;
  unsigned short* wq = p;     p += (size_t)Df * Df;       // 2MB
  unsigned short* wk = p;     p += (size_t)Df * Df;
  unsigned short* wo = p;     p += (size_t)Df * Df;
  unsigned short* textb = p;  p += (size_t)Bsz * Tt * Df; // 8MB
  unsigned short* featsb = p; p += (size_t)Bsz * Nn * Df; // 16MB
  unsigned short* vtb = p;    p += (size_t)Bsz * Df * Nn; // 16MB
  unsigned short* qb = p;     p += (size_t)Bsz * Tt * Df; // 8MB
  unsigned short* kb = p;     p += (size_t)Bsz * Nn * Df; // 16MB
  unsigned short* aob = p;    p += (size_t)Bsz * Tt * Df; // 8MB

  const int nW = Df * Df;
  const int nT = Bsz * Tt * Df;
  cvt_bf16<<<nW / 256, 256, 0, stream>>>(W_q, wq, nW);
  cvt_bf16<<<nW / 256, 256, 0, stream>>>(W_k, wk, nW);
  cvt_bf16<<<nW / 256, 256, 0, stream>>>(W_o, wo, nW);
  cvt_bf16<<<nT / 256, 256, 0, stream>>>(text, textb, nT);

  rmsnorm_feat<<<Bsz * Nn, 256, 0, stream>>>(feat, g_feat, featsb, vtb);

  gemm16_rms<<<Bsz * Tt / 32, 256, 0, stream>>>(textb, wq, b_q, g_q, qb);
  gemm16_rms<<<Bsz * Nn / 32, 256, 0, stream>>>(featsb, wk, b_k, g_k, kb);

  attention<<<dim3(Tt / 64, Hh, Bsz), 128, 0, stream>>>(qb, kb, vtb, aob);

  gemm16_bias<<<Bsz * Tt / 32, 256, 0, stream>>>(aob, wo, b_o, out);
}
